// MultiHeadCausalAttention_43490838840038
// MI455X (gfx1250) — compile-verified, matrix operand order checked by reading, not run on hardware
//
#include <hip/hip_runtime.h>
#include <hip/hip_bf16.h>
#include <stdint.h>

typedef __attribute__((ext_vector_type(16))) _Float16 v16h;
typedef __attribute__((ext_vector_type(8)))  float    v8f;
typedef __attribute__((ext_vector_type(8)))  uint32_t v8u;

#define B_NUM 2
#define S_LEN 2048
#define E_DIM 1024
#define A_DIM 1024
#define H_NUM 16
#define D_DIM 64

// ---------------------------------------------------------------------------
// WMMA helpers (wave32, 16x16x32 f16 -> f32).  Fragment gathers are 8
// dwords per lane in two contiguous runs of 4 (offsets 0-3, 8-11) (lowered to 2x global_load_b128).
// ---------------------------------------------------------------------------
__device__ inline v16h frag_load(const _Float16* __restrict__ base, int ld,
                                 int rc, int k0, int lane) {
  const uint32_t* p = reinterpret_cast<const uint32_t*>(
      base + (size_t)rc * (size_t)ld + (size_t)(k0 + ((lane & 16) >> 1)));
  v8u u;
#pragma unroll
  for (int i = 0; i < 8; ++i) {
    int idx = (i & 3) + ((i >> 2) << 3);   // dword offsets 0,1,2,3, 8,9,10,11
    u[i] = p[idx];
  }
  return __builtin_bit_cast(v16h, u);
}

__device__ inline v8f wmma_f16(v16h a, v16h b, v8f c) {
  return __builtin_amdgcn_wmma_f32_16x16x32_f16(false, a, false, b,
                                                (short)0, c, false, false);
}

// ---------------------------------------------------------------------------
// Precision conversion / weight transpose (tiny vs. 537MB attn I/O)
// ---------------------------------------------------------------------------
__global__ void cvt_f32_to_f16(const float* __restrict__ src,
                               _Float16* __restrict__ dst, int n) {
  int i = blockIdx.x * blockDim.x + threadIdx.x;
  if (i < n) dst[i] = (_Float16)src[i];
}

// W[k*N + n] (f32) -> Wt[n*K + k] (f16): makes B-fragment loads contiguous.
__global__ void transpose_to_f16(const float* __restrict__ W,
                                 _Float16* __restrict__ Wt, int K, int N) {
  int i = blockIdx.x * blockDim.x + threadIdx.x;
  if (i < K * N) {
    int k = i / N, n = i % N;
    Wt[(size_t)n * K + k] = (_Float16)W[i];
  }
}

// ---------------------------------------------------------------------------
// Fused QKV projection, software-pipelined. grid = (M/16, A/64, 3), block=32.
// z=0: Q (pre-scaled by 1/sqrt(D)) -> Qh[b][h][s][d]
// z=1: K                           -> Kh[b][h][s][d]
// z=2: V stored transposed         -> Vt[b][h][d][s]
// ---------------------------------------------------------------------------
__global__ void qkv_gemm(const _Float16* __restrict__ x16,
                         const _Float16* __restrict__ WqT,
                         const _Float16* __restrict__ WkT,
                         const _Float16* __restrict__ WvT,
                         const float* __restrict__ bq,
                         const float* __restrict__ bk,
                         const float* __restrict__ bv,
                         _Float16* __restrict__ Qh,
                         _Float16* __restrict__ Kh,
                         _Float16* __restrict__ Vt) {
  const int lane = threadIdx.x;
  const int tm = blockIdx.x, tn = blockIdx.y, z = blockIdx.z;
  const _Float16* Wt = (z == 0) ? WqT : (z == 1) ? WkT : WvT;
  const float* bias  = (z == 0) ? bq  : (z == 1) ? bk  : bv;

  v8f acc[4] = {};
  const int row = tm * 16 + (lane & 15);
  const int n0 = tn * 64;

  // prologue: fragments for k0 = 0
  v16h a = frag_load(x16, E_DIM, row, 0, lane);
  v16h bb[4];
#pragma unroll
  for (int t = 0; t < 4; ++t)
    bb[t] = frag_load(Wt, E_DIM, n0 + t * 16 + (lane & 15), 0, lane);

  for (int k0 = 0; k0 < E_DIM; k0 += 32) {
    // issue NEXT iteration's loads before consuming current fragments:
    // the wait in front of the WMMAs then only drains past these.
    const int kn = (k0 + 32 < E_DIM) ? k0 + 32 : k0;
    v16h an = frag_load(x16, E_DIM, row, kn, lane);
    v16h bn[4];
#pragma unroll
    for (int t = 0; t < 4; ++t)
      bn[t] = frag_load(Wt, E_DIM, n0 + t * 16 + (lane & 15), kn, lane);
    if (k0 + 64 < E_DIM) {   // near-scope prefetch one more tile ahead
      __builtin_prefetch(x16 + (size_t)row * E_DIM + k0 + 64, 0, 3);
      __builtin_prefetch(Wt + (size_t)(n0 + (lane & 15)) * E_DIM + k0 + 64, 0, 3);
    }
#pragma unroll
    for (int t = 0; t < 4; ++t)
      acc[t] = wmma_f16(a, bb[t], acc[t]);
    a = an;
#pragma unroll
    for (int t = 0; t < 4; ++t) bb[t] = bn[t];
  }

  const int hi8 = (lane & 16) >> 1;
#pragma unroll
  for (int t = 0; t < 4; ++t) {
    int col = n0 + t * 16 + (lane & 15);
    float bval = bias[col];
    int h = col >> 6, d = col & 63;
#pragma unroll
    for (int r = 0; r < 8; ++r) {
      int m = tm * 16 + r + hi8;
      int b = m >> 11, s = m & (S_LEN - 1);
      size_t bh = (size_t)(b * H_NUM + h);
      float v = acc[t][r] + bval;
      if (z == 0)
        Qh[(bh * S_LEN + s) * D_DIM + d] = (_Float16)(v * 0.125f);  // 1/sqrt(64)
      else if (z == 1)
        Kh[(bh * S_LEN + s) * D_DIM + d] = (_Float16)v;
      else
        Vt[(bh * D_DIM + d) * S_LEN + s] = (_Float16)v;
    }
  }
}

// ---------------------------------------------------------------------------
// Fused causal attention, 4 waves (128 threads) per 16-row q-tile.
// grid = (S/16, B*H).  LDS: 16x2048 f32 scores (128KB) + 4x16x64 f32 ctx
// partials (16KB) + 32 row stats => ~144KB -> 2 blocks (8 waves) per WGP.
// Waves split score j-tiles and ctx k-chunks; softmax/attn-write is
// row-serial with lane-contiguous columns for coalesced 512B bursts.
// ---------------------------------------------------------------------------
__global__ void attn_kernel(const _Float16* __restrict__ Qh,
                            const _Float16* __restrict__ Kh,
                            const _Float16* __restrict__ Vt,
                            float* __restrict__ attn_out,
                            _Float16* __restrict__ ctx16) {
  extern __shared__ float smem[];
  float* sc      = smem;                       // [16][S_LEN]
  float* red     = smem + 16 * S_LEN;          // [4][16][64]
  float* rowstat = red + 4 * 16 * 64;          // [2][16] : max, 1/sum

  const float NEG_INF = -__builtin_inff();
  const int tid  = threadIdx.x;                // 0..127
  const int lane = tid & 31;
  const int w    = tid >> 5;                   // wave id 0..3
  const int qt = blockIdx.x, bh = blockIdx.y;
  const int q0 = qt * 16;
  const int jend = (qt + 1) * 16;
  const int jend32 = (jend + 31) & ~31;

  const _Float16* Qb = Qh + (size_t)bh * S_LEN * D_DIM;
  const _Float16* Kb = Kh + (size_t)bh * S_LEN * D_DIM;
  const _Float16* Vb = Vt + (size_t)bh * D_DIM * S_LEN;

  const int nl = lane & 15;
  const int hi8 = (lane & 16) >> 1;

  // Q fragments (D=64 -> two K-halves); same in every wave
  v16h qa0 = frag_load(Qb, D_DIM, q0 + nl, 0, lane);
  v16h qa1 = frag_load(Qb, D_DIM, q0 + nl, 32, lane);

  // ---- scores: wave w handles j = w, w+4, ... (double-buffered K) ----
  if (w <= qt) {
    v16h kb0 = frag_load(Kb, D_DIM, w * 16 + nl, 0, lane);
    v16h kb1 = frag_load(Kb, D_DIM, w * 16 + nl, 32, lane);
    for (int j = w; j <= qt; j += 4) {
      const int jn = (j + 4 <= qt) ? j + 4 : j;
      v16h kn0 = frag_load(Kb, D_DIM, jn * 16 + nl, 0, lane);
      v16h kn1 = frag_load(Kb, D_DIM, jn * 16 + nl, 32, lane);
      if (j + 8 <= qt)
        __builtin_prefetch(Kb + (size_t)((j + 8) * 16 + nl) * D_DIM, 0, 3);
      v8f s = {};
      s = wmma_f16(qa0, kb0, s);
      s = wmma_f16(qa1, kb1, s);
      kb0 = kn0; kb1 = kn1;
      int col = j * 16 + nl;
#pragma unroll
      for (int r = 0; r < 8; ++r) {
        int m = r + hi8;
        sc[m * S_LEN + col] = (col <= q0 + m) ? s[r] : NEG_INF;
      }
    }
  }
  __syncthreads();

  // ---- row stats: 8 contiguous lanes per row (tid = r*8 + i) ----
  {
    const int r = tid >> 3, i = tid & 7;
    float mx = NEG_INF;
    for (int c = i; c < jend; c += 8) mx = fmaxf(mx, sc[r * S_LEN + c]);
#pragma unroll
    for (int d = 1; d < 8; d <<= 1) mx = fmaxf(mx, __shfl_xor(mx, d, 32));
    float sum = 0.f;
    for (int c = i; c < jend; c += 8) sum += __expf(sc[r * S_LEN + c] - mx);
#pragma unroll
    for (int d = 1; d < 8; d <<= 1) sum += __shfl_xor(sum, d, 32);
    if (i == 0) { rowstat[r] = mx; rowstat[16 + r] = 1.f / sum; }
  }
  __syncthreads();

  // ---- attn written ONCE, coalesced; probs overwritten into LDS ----
  for (int r = 0; r < 16; ++r) {
    const float mx = rowstat[r], inv = rowstat[16 + r];
    float* arow = attn_out + ((size_t)bh * S_LEN + q0 + r) * S_LEN;
    for (int c = tid; c < S_LEN; c += 128) {
      float p = 0.f;
      if (c < jend) {
        p = __expf(sc[r * S_LEN + c] - mx) * inv;   // masked cols: exp(-inf)=0
        sc[r * S_LEN + c] = p;
      }
      arow[c] = p;
    }
  }
  // zero-pad [jend, jend32) so the last 32-wide WMMA K chunk is clean
  for (int idx = tid; idx < 16 * (jend32 - jend); idx += 128) {
    sc[(idx & 15) * S_LEN + jend + (idx >> 4)] = 0.f;
  }
  __syncthreads();

  // ---- ctx = P @ V: wave w takes k-chunks w*32, w*32+128, ... ----
  v8f acc[4] = {};
  const float* prow = sc + (size_t)nl * S_LEN;
  if (w * 32 < jend32) {
    v16h bb[4];
#pragma unroll
    for (int t = 0; t < 4; ++t)
      bb[t] = frag_load(Vb, S_LEN, t * 16 + nl, w * 32, lane);
    for (int k0 = w * 32; k0 < jend32; k0 += 128) {
      const int kn = (k0 + 128 < jend32) ? k0 + 128 : k0;
      v16h bn[4];
#pragma unroll
      for (int t = 0; t < 4; ++t)
        bn[t] = frag_load(Vb, S_LEN, t * 16 + nl, kn, lane);
      // A fragment from f32 probs in LDS (convert on the fly)
      v16h a = {};
#pragma unroll
      for (int i = 0; i < 8; ++i) {
        int kk = k0 + ((i & 3) << 1) + ((i >> 2) << 4) + ((lane & 16) >> 1);
        a[2 * i]     = (_Float16)prow[kk];
        a[2 * i + 1] = (_Float16)prow[kk + 1];
      }
#pragma unroll
      for (int t = 0; t < 4; ++t)
        acc[t] = wmma_f16(a, bb[t], acc[t]);
#pragma unroll
      for (int t = 0; t < 4; ++t) bb[t] = bn[t];
    }
  }
  // stash per-wave partials
#pragma unroll
  for (int t = 0; t < 4; ++t)
#pragma unroll
    for (int r = 0; r < 8; ++r)
      red[w * 1024 + (r + hi8) * 64 + t * 16 + nl] = acc[t][r];
  __syncthreads();

  // ---- cross-wave reduce + merge heads: ctx16[b][s][h*64+d] (f16) ----
  const int b_idx = bh / H_NUM, h_idx = bh % H_NUM;
  for (int e = tid; e < 16 * 64; e += 128) {
    float v = red[e] + red[1024 + e] + red[2048 + e] + red[3072 + e];
    int s = q0 + (e >> 6), d = e & 63;
    ctx16[((size_t)b_idx * S_LEN + s) * A_DIM + h_idx * D_DIM + d] = (_Float16)v;
  }
}

// ---------------------------------------------------------------------------
// Output projection, software-pipelined: out = ctx @ Wo + bo (fp32 result).
// grid=(M/16, E/64), block = 32.
// ---------------------------------------------------------------------------
__global__ void out_gemm(const _Float16* __restrict__ ctx16,
                         const _Float16* __restrict__ WoT,
                         const float* __restrict__ bo,
                         float* __restrict__ out) {
  const int lane = threadIdx.x;
  const int tm = blockIdx.x, tn = blockIdx.y;
  v8f acc[4] = {};
  const int row = tm * 16 + (lane & 15);
  const int n0 = tn * 64;

  v16h a = frag_load(ctx16, A_DIM, row, 0, lane);
  v16h bb[4];
#pragma unroll
  for (int t = 0; t < 4; ++t)
    bb[t] = frag_load(WoT, A_DIM, n0 + t * 16 + (lane & 15), 0, lane);

  for (int k0 = 0; k0 < A_DIM; k0 += 32) {
    const int kn = (k0 + 32 < A_DIM) ? k0 + 32 : k0;
    v16h an = frag_load(ctx16, A_DIM, row, kn, lane);
    v16h bn[4];
#pragma unroll
    for (int t = 0; t < 4; ++t)
      bn[t] = frag_load(WoT, A_DIM, n0 + t * 16 + (lane & 15), kn, lane);
    if (k0 + 64 < A_DIM) {
      __builtin_prefetch(ctx16 + (size_t)row * A_DIM + k0 + 64, 0, 3);
      __builtin_prefetch(WoT + (size_t)(n0 + (lane & 15)) * A_DIM + k0 + 64, 0, 3);
    }
#pragma unroll
    for (int t = 0; t < 4; ++t)
      acc[t] = wmma_f16(a, bb[t], acc[t]);
    a = an;
#pragma unroll
    for (int t = 0; t < 4; ++t) bb[t] = bn[t];
  }

  const int hi8 = (lane & 16) >> 1;
#pragma unroll
  for (int t = 0; t < 4; ++t) {
    int col = n0 + t * 16 + (lane & 15);
    float bias = bo[col];
#pragma unroll
    for (int r = 0; r < 8; ++r) {
      int mrow = tm * 16 + r + hi8;
      out[(size_t)mrow * E_DIM + col] = acc[t][r] + bias;
    }
  }
}

// ---------------------------------------------------------------------------
extern "C" void kernel_launch(void* const* d_in, const int* in_sizes, int n_in,
                              void* d_out, int out_size, void* d_ws, size_t ws_size,
                              hipStream_t stream) {
  const float* x  = (const float*)d_in[0];
  const float* Wq = (const float*)d_in[1];
  const float* bq = (const float*)d_in[2];
  const float* Wk = (const float*)d_in[3];
  const float* bk = (const float*)d_in[4];
  const float* Wv = (const float*)d_in[5];
  const float* bv = (const float*)d_in[6];
  const float* Wo = (const float*)d_in[7];
  const float* bo = (const float*)d_in[8];

  const size_t M = (size_t)B_NUM * S_LEN;            // 4096 rows
  _Float16* ws = (_Float16*)d_ws;
  size_t off = 0;
  _Float16* x16  = ws + off; off += M * E_DIM;
  _Float16* WqT  = ws + off; off += (size_t)E_DIM * A_DIM;
  _Float16* WkT  = ws + off; off += (size_t)E_DIM * A_DIM;
  _Float16* WvT  = ws + off; off += (size_t)E_DIM * A_DIM;
  _Float16* WoT  = ws + off; off += (size_t)A_DIM * E_DIM;
  _Float16* Qh   = ws + off; off += M * A_DIM;
  _Float16* Kh   = ws + off; off += M * A_DIM;
  _Float16* Vt   = ws + off; off += M * A_DIM;
  _Float16* ctx16= ws + off; off += M * A_DIM;       // ~48 MB total

  float* out  = (float*)d_out;
  float* attn = out + M * E_DIM;                     // tuple: (out, attn)

  int nX = (int)(M * E_DIM);
  cvt_f32_to_f16<<<(nX + 255) / 256, 256, 0, stream>>>(x, x16, nX);
  int nW = E_DIM * A_DIM;
  int gW = (nW + 255) / 256;
  transpose_to_f16<<<gW, 256, 0, stream>>>(Wq, WqT, E_DIM, A_DIM);
  transpose_to_f16<<<gW, 256, 0, stream>>>(Wk, WkT, E_DIM, A_DIM);
  transpose_to_f16<<<gW, 256, 0, stream>>>(Wv, WvT, E_DIM, A_DIM);
  transpose_to_f16<<<gW, 256, 0, stream>>>(Wo, WoT, A_DIM, E_DIM);

  dim3 g1((unsigned)(M / 16), A_DIM / 64, 3);
  qkv_gemm<<<g1, 32, 0, stream>>>(x16, WqT, WkT, WvT, bq, bk, bv, Qh, Kh, Vt);

  // LDS: scores + ctx partials + row stats
  size_t smem_bytes = (16 * S_LEN + 4 * 16 * 64 + 32) * sizeof(float);
  dim3 g2(S_LEN / 16, B_NUM * H_NUM);
  attn_kernel<<<g2, 128, smem_bytes, stream>>>(Qh, Kh, Vt, attn, ctx16);

  dim3 g3((unsigned)(M / 16), E_DIM / 64);
  out_gemm<<<g3, 32, 0, stream>>>(ctx16, WoT, bo, out);
}
